// MultHeadSelfAttention_88862873355025
// MI455X (gfx1250) — matrix operand order repaired and checked by reading, not compiled, not run on hardware
//
#include <hip/hip_runtime.h>
#include <hip/hip_bf16.h>

typedef __attribute__((ext_vector_type(16))) _Float16 v16h;
typedef __attribute__((ext_vector_type(8)))  float    v8f;

#define DIM_IN   512
#define NHEADS   32
#define DK       64
#define SEQ      2048
#define BATCH    2
#define MROWS    (BATCH * SEQ)        // 4096
#define INNER    (NHEADS * DK)        // 2048

// ---------------------------------------------------------------------------
// WMMA + async-copy helpers (gfx1250 wave32)
// ---------------------------------------------------------------------------
__device__ __forceinline__ v8f wmma_f16(v16h a, v16h b, v8f c) {
  return __builtin_amdgcn_wmma_f32_16x16x32_f16(false, a, false, b, (short)0, c,
                                                false, false);
}

// Async global->LDS 16-byte copy (ASYNCcnt-tracked, bypasses VGPRs).
// VDST holds the wave-relative LDS byte offset = low 32 bits of the generic
// address of a __shared__ object; VADDR is the 64-bit global address.
__device__ __forceinline__ void async_ld_b128(void* lds_ptr, const void* gptr) {
  unsigned l = (unsigned)(size_t)lds_ptr;
  asm volatile("global_load_async_to_lds_b128 %0, %1, off"
               :: "v"(l), "v"(gptr) : "memory");
}
__device__ __forceinline__ void wait_async0() {
  asm volatile("s_wait_asynccnt 0" ::: "memory");
}

// A fragment: 16x32 f16, row-major source, leading dim `ld` (halves).
// VGPR j holds K = khalf*16 + 2*j .. +1 (consecutive pair).
__device__ __forceinline__ v16h load_a_frag(const _Float16* base, int ld, int lane) {
  const _Float16* row = base + (size_t)(lane & 15) * ld;
  int kh = ((lane >> 4) & 1) * 16;
  v16h f;
#pragma unroll
  for (int j = 0; j < 8; ++j) {
    int k = kh + (j << 1);
    f[2 * j]     = row[k];
    f[2 * j + 1] = row[k + 1];
  }
  return f;
}

// B fragment from an n-major source bt[n][k], leading dim `ld` (halves).
// Lane holds column n = lane&15, K = (lane>>4)*16 + e -> 16 contiguous halves.
__device__ __forceinline__ v16h load_bt_frag(const _Float16* base, int ld, int lane) {
  const _Float16* p = base + (size_t)(lane & 15) * ld + ((lane >> 4) << 4);
  return *(const v16h*)p;
}

// ---------------------------------------------------------------------------
// Stage 0: conversions. cvt: straight f32->f16. cvt_tr: transpose so GEMM
// B-tiles become contiguous-row copies (async-copy friendly): out[n][k]=in[k][n].
// ---------------------------------------------------------------------------
__global__ void cvt_f32_f16(const float* __restrict__ in, _Float16* __restrict__ out,
                            int n) {
  int i = blockIdx.x * 1024 + threadIdx.x;
#pragma unroll
  for (int j = 0; j < 4; ++j) {
    int idx = i + j * 256;
    if (idx < n) out[idx] = (_Float16)in[idx];
  }
}

__global__ void cvt_tr_f32_f16(const float* __restrict__ in, _Float16* __restrict__ out,
                               int Nd, int Kd) {
  int idx = blockIdx.x * 256 + threadIdx.x;
  if (idx < Nd * Kd) {
    int n = idx / Kd, k = idx - n * Kd;
    out[idx] = (_Float16)in[(size_t)k * Nd + n];
  }
}

// ---------------------------------------------------------------------------
// Stage 1: QKV projection  x[4096x512] @ W  (W given as W^T [2048][512]).
// vlayout 0: Out[((b*32+h)*2048+n)*64+d]  (Q, K; Q pre-scaled by 1/8)
// vlayout 1: Out[((b*32+h)*64+d)*2048+n]  (V stored d-major for async staging)
// 256 thr / 8 waves; tile 128x64, BK=32; double-buffered async LDS staging.
// ---------------------------------------------------------------------------
__global__ void gemm_qkv(const _Float16* __restrict__ A, const _Float16* __restrict__ Bt,
                         _Float16* __restrict__ Out, float scale, int vlayout) {
  __shared__ alignas(64) _Float16 sA[2][128 * 32];
  __shared__ alignas(64) _Float16 sB[2][64 * 32];
  const int tid = threadIdx.x, lane = tid & 31, wave = tid >> 5;
  const int bm = blockIdx.y * 128, bn = blockIdx.x * 64;
  const int wm = (wave & 3) * 32, wn = (wave >> 2) * 32;

  auto prefetch = [&](int buf, int kt) {
#pragma unroll
    for (int i = 0; i < 2; ++i) {              // A tile: 128 rows x 64B
      int c = tid + i * 256;
      int row = c >> 2, seg = c & 3;
      async_ld_b128(&sA[buf][row * 32 + seg * 8],
                    A + (size_t)(bm + row) * DIM_IN + kt + seg * 8);
    }
    {                                          // B tile: 64 rows x 64B
      int row = tid >> 2, seg = tid & 3;
      async_ld_b128(&sB[buf][row * 32 + seg * 8],
                    Bt + (size_t)(bn + row) * DIM_IN + kt + seg * 8);
    }
  };

  v8f acc[2][2] = {};
  prefetch(0, 0);
  const int NK = DIM_IN / 32;                  // 16
  for (int it = 0; it < NK; ++it) {
    wait_async0();
    __syncthreads();
    int buf = it & 1;
    if (it + 1 < NK) prefetch(buf ^ 1, (it + 1) * 32);

    v16h a0 = load_a_frag(sA[buf] + wm * 32, 32, lane);
    v16h a1 = load_a_frag(sA[buf] + (wm + 16) * 32, 32, lane);
    v16h b0 = load_bt_frag(sB[buf] + wn * 32, 32, lane);
    v16h b1 = load_bt_frag(sB[buf] + (wn + 16) * 32, 32, lane);
    acc[0][0] = wmma_f16(a0, b0, acc[0][0]);
    acc[0][1] = wmma_f16(a0, b1, acc[0][1]);
    acc[1][0] = wmma_f16(a1, b0, acc[1][0]);
    acc[1][1] = wmma_f16(a1, b1, acc[1][1]);
  }

  const int r0 = (lane >> 4) * 8, nn = lane & 15;
#pragma unroll
  for (int mi = 0; mi < 2; ++mi)
#pragma unroll
    for (int ni = 0; ni < 2; ++ni)
#pragma unroll
      for (int r = 0; r < 8; ++r) {
        int m = bm + wm + mi * 16 + r0 + r;    // [0,4096)
        int c = bn + wn + ni * 16 + nn;        // [0,2048)
        int b = m >> 11, nrow = m & (SEQ - 1);
        int h = c >> 6, d = c & 63;
        size_t bh = (size_t)b * NHEADS + h;
        size_t off = vlayout ? (bh * DK + d) * SEQ + nrow
                             : ((bh * SEQ + nrow) << 6) + d;
        Out[off] = (_Float16)(acc[mi][ni][r] * scale);
      }
}

// ---------------------------------------------------------------------------
// Stage 2: flash attention. Grid (B*H, SEQ/128); 8 waves x 16 query rows.
// K chunk [32xDk] and Vt chunk [Dkx32] double-buffered via async copies.
// ---------------------------------------------------------------------------
__global__ void attn(const _Float16* __restrict__ Q, const _Float16* __restrict__ K,
                     const _Float16* __restrict__ Vt, _Float16* __restrict__ Outp) {
  __shared__ alignas(64) _Float16 sK[2][32 * 64];   // [key][d]
  __shared__ alignas(64) _Float16 sVt[2][64 * 32];  // [d][key]
  __shared__ alignas(64) _Float16 sP[8][16 * 32];   // per-wave P tile

  const int tid = threadIdx.x, lane = tid & 31, wave = tid >> 5;
  const int bh = blockIdx.x;                        // 0..63
  const int q0 = blockIdx.y * 128 + wave * 16;
  const size_t base = (size_t)bh * SEQ * DK;        // same for Q/K and Vt

  v16h qa0 = load_a_frag(Q + base + (size_t)q0 * DK, DK, lane);
  v16h qa1 = load_a_frag(Q + base + (size_t)q0 * DK + 32, DK, lane);

  auto prefetch = [&](int buf, int kc) {
    {                                              // K: 32 rows x 128B
      int row = tid >> 3, seg = tid & 7;
      async_ld_b128(&sK[buf][row * 64 + seg * 8],
                    K + base + (size_t)(kc + row) * DK + seg * 8);
    }
    {                                              // Vt: 64 rows x 64B
      int row = tid >> 2, seg = tid & 3;
      async_ld_b128(&sVt[buf][row * 32 + seg * 8],
                    Vt + base + (size_t)row * SEQ + kc + seg * 8);
    }
  };

  v8f o0 = {}, o1 = {}, o2 = {}, o3 = {};
  float mrow[8], lrow[8];
#pragma unroll
  for (int r = 0; r < 8; ++r) { mrow[r] = -1e30f; lrow[r] = 0.f; }

  prefetch(0, 0);
  const int NC = SEQ / 32;                         // 64 chunks
  for (int it = 0; it < NC; ++it) {
    wait_async0();
    __syncthreads();
    int buf = it & 1;
    if (it + 1 < NC) prefetch(buf ^ 1, (it + 1) * 32);

    // S = Q K^T (sK rows = keys, contiguous in d -> B^T layout)
    v8f s0 = {}, s1 = {};
    s0 = wmma_f16(qa0, load_bt_frag(sK[buf], DK, lane), s0);
    s0 = wmma_f16(qa1, load_bt_frag(sK[buf] + 32, DK, lane), s0);
    s1 = wmma_f16(qa0, load_bt_frag(sK[buf] + 16 * DK, DK, lane), s1);
    s1 = wmma_f16(qa1, load_bt_frag(sK[buf] + 16 * DK + 32, DK, lane), s1);

    // Online softmax; row m = (lane>>4)*8+r lives on 16 lanes -> xor reductions
    float alpha[8];
#pragma unroll
    for (int r = 0; r < 8; ++r) {
      float mx = fmaxf(s0[r], s1[r]);
#pragma unroll
      for (int off = 1; off < 16; off <<= 1) mx = fmaxf(mx, __shfl_xor(mx, off, 32));
      float mnew = fmaxf(mrow[r], mx);
      alpha[r] = __expf(mrow[r] - mnew);
      float p0 = __expf(s0[r] - mnew);
      float p1 = __expf(s1[r] - mnew);
      float sum = p0 + p1;
#pragma unroll
      for (int off = 1; off < 16; off <<= 1) sum += __shfl_xor(sum, off, 32);
      lrow[r] = lrow[r] * alpha[r] + sum;
      mrow[r] = mnew;
      s0[r] = p0; s1[r] = p1;
    }
#pragma unroll
    for (int r = 0; r < 8; ++r) {
      o0[r] *= alpha[r]; o1[r] *= alpha[r]; o2[r] *= alpha[r]; o3[r] *= alpha[r];
    }

    // Reshape P C-layout -> A-layout through per-wave LDS
    _Float16* p = sP[wave];
    {
      int m = (lane >> 4) * 8, n = lane & 15;
#pragma unroll
      for (int r = 0; r < 8; ++r) {
        p[(m + r) * 32 + n]      = (_Float16)s0[r];
        p[(m + r) * 32 + 16 + n] = (_Float16)s1[r];
      }
    }
    asm volatile("s_wait_dscnt 0" ::: "memory");   // same-wave DS store->load

    v16h pa = load_a_frag(p, 32, lane);
    o0 = wmma_f16(pa, load_bt_frag(sVt[buf],           32, lane), o0);
    o1 = wmma_f16(pa, load_bt_frag(sVt[buf] + 16 * 32, 32, lane), o1);
    o2 = wmma_f16(pa, load_bt_frag(sVt[buf] + 32 * 32, 32, lane), o2);
    o3 = wmma_f16(pa, load_bt_frag(sVt[buf] + 48 * 32, 32, lane), o3);
  }

  // Normalize; att stored f16 as [B, N, H*64] for the final GEMM.
  const int b = bh >> 5, h = bh & 31;
  const int r0 = (lane >> 4) * 8, nn = lane & 15;
#pragma unroll
  for (int r = 0; r < 8; ++r) {
    float inv = 1.0f / lrow[r];
    int nq = q0 + r0 + r;
    size_t ob = ((size_t)b * SEQ + nq) * INNER + (size_t)h * DK;
    Outp[ob + 0  + nn] = (_Float16)(o0[r] * inv);
    Outp[ob + 16 + nn] = (_Float16)(o1[r] * inv);
    Outp[ob + 32 + nn] = (_Float16)(o2[r] * inv);
    Outp[ob + 48 + nn] = (_Float16)(o3[r] * inv);
  }
}

// ---------------------------------------------------------------------------
// Stage 3: att[4096x2048] @ Wz (given as Wz^T [512][2048]) + bz -> f32 d_out
// ---------------------------------------------------------------------------
__global__ void gemm_out(const _Float16* __restrict__ A, const _Float16* __restrict__ Bt,
                         const float* __restrict__ bias, float* __restrict__ Out) {
  __shared__ alignas(64) _Float16 sA[2][128 * 32];
  __shared__ alignas(64) _Float16 sB[2][64 * 32];
  const int tid = threadIdx.x, lane = tid & 31, wave = tid >> 5;
  const int bm = blockIdx.y * 128, bn = blockIdx.x * 64;
  const int wm = (wave & 3) * 32, wn = (wave >> 2) * 32;

  auto prefetch = [&](int buf, int kt) {
#pragma unroll
    for (int i = 0; i < 2; ++i) {
      int c = tid + i * 256;
      int row = c >> 2, seg = c & 3;
      async_ld_b128(&sA[buf][row * 32 + seg * 8],
                    A + (size_t)(bm + row) * INNER + kt + seg * 8);
    }
    {
      int row = tid >> 2, seg = tid & 3;
      async_ld_b128(&sB[buf][row * 32 + seg * 8],
                    Bt + (size_t)(bn + row) * INNER + kt + seg * 8);
    }
  };

  v8f acc[2][2] = {};
  prefetch(0, 0);
  const int NK = INNER / 32;                   // 64
  for (int it = 0; it < NK; ++it) {
    wait_async0();
    __syncthreads();
    int buf = it & 1;
    if (it + 1 < NK) prefetch(buf ^ 1, (it + 1) * 32);

    v16h a0 = load_a_frag(sA[buf] + wm * 32, 32, lane);
    v16h a1 = load_a_frag(sA[buf] + (wm + 16) * 32, 32, lane);
    v16h b0 = load_bt_frag(sB[buf] + wn * 32, 32, lane);
    v16h b1 = load_bt_frag(sB[buf] + (wn + 16) * 32, 32, lane);
    acc[0][0] = wmma_f16(a0, b0, acc[0][0]);
    acc[0][1] = wmma_f16(a0, b1, acc[0][1]);
    acc[1][0] = wmma_f16(a1, b0, acc[1][0]);
    acc[1][1] = wmma_f16(a1, b1, acc[1][1]);
  }

  const int r0 = (lane >> 4) * 8, nn = lane & 15;
#pragma unroll
  for (int mi = 0; mi < 2; ++mi)
#pragma unroll
    for (int ni = 0; ni < 2; ++ni)
#pragma unroll
      for (int r = 0; r < 8; ++r) {
        int m = bm + wm + mi * 16 + r0 + r;
        int c = bn + wn + ni * 16 + nn;
        Out[(size_t)m * DIM_IN + c] = acc[mi][ni][r] + bias[c];
      }
}

// ---------------------------------------------------------------------------
extern "C" void kernel_launch(void* const* d_in, const int* in_sizes, int n_in,
                              void* d_out, int out_size, void* d_ws, size_t ws_size,
                              hipStream_t stream) {
  const float* x  = (const float*)d_in[0];
  const float* Wq = (const float*)d_in[1];
  const float* Wk = (const float*)d_in[2];
  const float* Wv = (const float*)d_in[3];
  const float* Wz = (const float*)d_in[4];
  const float* bz = (const float*)d_in[5];

  _Float16* ws = (_Float16*)d_ws;
  const size_t nX = (size_t)MROWS * DIM_IN;     // 2,097,152
  const size_t nW = (size_t)DIM_IN * INNER;     // 1,048,576
  const size_t nQ = (size_t)MROWS * INNER;      // 8,388,608
  _Float16* x16   = ws;
  _Float16* wqT   = x16 + nX;
  _Float16* wkT   = wqT + nW;
  _Float16* wvT   = wkT + nW;
  _Float16* wzT   = wvT + nW;
  _Float16* q16   = wzT + nW;
  _Float16* k16   = q16 + nQ;
  _Float16* vT16  = k16 + nQ;
  _Float16* att16 = vT16 + nQ;

  // Stage 0: conversions (weights transposed to N-major for async staging)
  cvt_f32_f16<<<(int)((nX + 1023) / 1024), 256, 0, stream>>>(x, x16, (int)nX);
  int gW = (int)((nW + 255) / 256);
  cvt_tr_f32_f16<<<gW, 256, 0, stream>>>(Wq, wqT, INNER, DIM_IN);
  cvt_tr_f32_f16<<<gW, 256, 0, stream>>>(Wk, wkT, INNER, DIM_IN);
  cvt_tr_f32_f16<<<gW, 256, 0, stream>>>(Wv, wvT, INNER, DIM_IN);
  cvt_tr_f32_f16<<<gW, 256, 0, stream>>>(Wz, wzT, DIM_IN, INNER);

  // Stage 1: QKV (Q pre-scaled by 1/sqrt(64); V stored d-major)
  dim3 gQKV(INNER / 64, MROWS / 128);           // 32 x 32
  gemm_qkv<<<gQKV, 256, 0, stream>>>(x16, wqT, q16, 0.125f, 0);
  gemm_qkv<<<gQKV, 256, 0, stream>>>(x16, wkT, k16, 1.0f, 0);
  gemm_qkv<<<gQKV, 256, 0, stream>>>(x16, wvT, vT16, 1.0f, 1);

  // Stage 2: flash attention
  dim3 gATT(BATCH * NHEADS, SEQ / 128);         // 64 x 16
  attn<<<gATT, 256, 0, stream>>>(q16, k16, vT16, att16);

  // Stage 3: output projection + bias -> f32
  dim3 gOUT(DIM_IN / 64, MROWS / 128);          // 8 x 32
  gemm_out<<<gOUT, 256, 0, stream>>>(att16, wzT, bz, (float*)d_out);
}
